// Generator_18124761989654
// MI455X (gfx1250) — matrix operand order repaired and checked by reading, not compiled, not run on hardware
//
#include <hip/hip_runtime.h>
#include <hip/hip_bf16.h>

#define B_ 16
#define L_ 64
#define E_ 256
#define V_ 16000
#define H_ 8
#define DH_ 32
#define O_ 2

#define ALDS_STRIDE 264  // 256 bf16 + 8 bf16 (16B) TDM pad per row -> conflict-free LDS

typedef __attribute__((ext_vector_type(16))) __bf16 bf16x16;
typedef __attribute__((ext_vector_type(8)))  __bf16 bf16x8;
typedef __attribute__((ext_vector_type(8)))  float  f32x8;
typedef __attribute__((ext_vector_type(4)))  unsigned int u32x4;
typedef __attribute__((ext_vector_type(8)))  int i32x8;
typedef __attribute__((ext_vector_type(4)))  int i32x4;

// ---------------------------------------------------------------------------
// Tensor Data Mover: async-copy a 2-D bf16 tile (nrows x rowElems, row stride
// strideElems) from global into LDS at ldsOff, adding 16B padding after every
// 512B row (pad_interval=6 -> 128 DWORDs, pad_amount=3 -> 4 DWORDs).
// D# layout per CDNA5 ISA 8.3/8.4. Issued by one wave; tracked by TENSORcnt.
// ---------------------------------------------------------------------------
__device__ __forceinline__ void tdm_load_2d_bf16(unsigned ldsOff,
                                                 const __bf16* gsrc,
                                                 unsigned rowElems,
                                                 unsigned nrows,
                                                 unsigned long long strideElems) {
  unsigned long long ga = (unsigned long long)gsrc;
  u32x4 g0;
  g0[0] = 1u;                                   // count=1, user, no gather
  g0[1] = ldsOff;                               // lds_addr (bytes)
  g0[2] = (unsigned)(ga & 0xffffffffu);         // global_addr[31:0]
  g0[3] = (unsigned)((ga >> 32) & 0x1ffffffu) | (2u << 30);  // addr[56:32] | type=2
  i32x8 g1;
  // wg_mask=0 | data_size=1(2B) | pad_enable | pad_interval=6 | pad_amount=3
  g1[0] = (int)((1u << 16) | (1u << 20) | (6u << 22) | (3u << 25));
  g1[1] = (int)((rowElems & 0xffffu) << 16);                 // tensor_dim0 lo
  g1[2] = (int)(((rowElems >> 16) & 0xffffu) | ((nrows & 0xffffu) << 16));  // dim0 hi | tensor_dim1 lo
  g1[3] = (int)(((nrows >> 16) & 0xffffu) | ((rowElems & 0xffffu) << 16));  // dim1 hi | tile_dim0
  g1[4] = (int)(nrows & 0xffffu);                            // tile_dim1 | tile_dim2=0
  g1[5] = (int)(strideElems & 0xffffffffull);                // tensor_dim0_stride lo
  g1[6] = (int)((strideElems >> 32) & 0xffffull);            // stride hi | dim1_stride lo=0
  g1[7] = 0;
  i32x4 z4 = {0, 0, 0, 0};
#if __has_include(<hip/amd_detail/amd_gfx1250_TDM.h>)
  i32x8 z8 = {0, 0, 0, 0, 0, 0, 0, 0};
  __builtin_amdgcn_tensor_load_to_lds(g0, g1, z4, z4, z8, 0);   // amdgpu-toolchain (6-arg)
#else
  __builtin_amdgcn_tensor_load_to_lds(g0, g1, z4, z4, 0);       // ROCm (5-arg)
#endif
}

// ---------------------------------------------------------------------------
// WMMA tile: C(16x16) = A(16x256 bf16, staged in LDS, padded rows) * B(256x16)
// BT is pre-transposed bf16 [N][K]; caller offsets BT by col0*256.
// ---------------------------------------------------------------------------
__device__ __forceinline__ f32x8 wmma_gemm_lds(const __bf16* __restrict__ Alds,
                                               const __bf16* __restrict__ BT,
                                               int lane) {
  const int m  = lane & 15;
  const int hi = lane >> 4;
  const __bf16* arow = Alds + m * ALDS_STRIDE;
  const __bf16* brow = BT + (long)m * 256;
  f32x8 acc = {};
#pragma unroll
  for (int k0 = 0; k0 < 256; k0 += 32) {
    // 16-bit A 16x32 layout: lane holds K=[hi*8..+7] and [16+hi*8..+7]
    bf16x8 a0 = *(const bf16x8*)(arow + k0 + hi * 8);
    bf16x8 a1 = *(const bf16x8*)(arow + k0 + hi * 8 + 16);
    bf16x16 af;
#pragma unroll
    for (int j = 0; j < 8; ++j) { af[j] = a0[j]; af[8 + j] = a1[j]; }
    // B: lane n holds K=[hi*8..+7] and [16+hi*8..+7] of column n
    bf16x8 b0 = *(const bf16x8*)(brow + k0 + hi * 8);
    bf16x8 b1 = *(const bf16x8*)(brow + k0 + hi * 8 + 16);
    bf16x16 bfv;
#pragma unroll
    for (int j = 0; j < 8; ++j) { bfv[j] = b0[j]; bfv[8 + j] = b1[j]; }
    acc = __builtin_amdgcn_wmma_f32_16x16x32_bf16(false, af, false, bfv,
                                                  (short)0, acc, false, false);
  }
  return acc;
}

__device__ __forceinline__ unsigned lds_offset(const void* p) {
  return (unsigned)(unsigned long long)p;  // low 32 bits of shared aperture = LDS offset
}

// ---------------------------------------------------------------------------
// Prep kernels (run once per launch)
// ---------------------------------------------------------------------------
__global__ void k_prep_pe_y(float* __restrict__ PE, float* __restrict__ Y,
                            __bf16* __restrict__ Ybf, int* __restrict__ out_tok) {
  int tid = blockIdx.x * 256 + threadIdx.x;
  if (tid < B_ * L_ * E_) { Y[tid] = 0.f; Ybf[tid] = (__bf16)0.f; }
  if (tid < L_ * E_) {
    int pos = tid / E_, e = tid % E_;
    int i2 = (e >> 1) << 1;
    float div = expf((float)i2 * (-logf(10000.f) / (float)E_));
    float a = (float)pos * div;
    PE[tid] = (e & 1) ? cosf(a) : sinf(a);
  }
  if (tid == 0) out_tok[0] = 0;  // START
}

__global__ void k_y0(float* __restrict__ Y, __bf16* __restrict__ Ybf,
                     const float* __restrict__ PE, const float* __restrict__ emb) {
  int tid = blockIdx.x * 256 + threadIdx.x;  // B_*E_
  if (tid >= B_ * E_) return;
  int b = tid / E_, e = tid % E_;
  float v = emb[e] + PE[e];  // emb[START=0] + PE[0]
  long o = ((long)b * L_) * E_ + e;
  Y[o] = v;
  Ybf[o] = (__bf16)v;
}

// Transpose+convert 8 ExE weight mats to bf16 [N][K]; order q0,q1,k0,k1,v0,v1,o0,o1
__global__ void k_trsq(const float* __restrict__ Wq, const float* __restrict__ Wk,
                       const float* __restrict__ Wv, const float* __restrict__ Wo,
                       __bf16* __restrict__ WT) {
  int tid = blockIdx.x * 256 + threadIdx.x;
  if (tid >= 8 * 65536) return;
  int mat = tid >> 16, rem = tid & 65535;
  int n = rem >> 8, k = rem & 255;
  int i = mat & 1;
  const float* src;
  switch (mat >> 1) {
    case 0: src = Wq; break;
    case 1: src = Wk; break;
    case 2: src = Wv; break;
    default: src = Wo; break;
  }
  WT[(long)mat * 65536 + n * 256 + k] = (__bf16)src[(long)i * 65536 + k * 256 + n];
}

__global__ void k_trsoft(const float* __restrict__ sW, __bf16* __restrict__ SWT) {
  long tid = (long)blockIdx.x * 256 + threadIdx.x;  // V_*E_
  if (tid >= (long)V_ * E_) return;
  int v = (int)(tid >> 8), k = (int)(tid & 255);
  SWT[tid] = (__bf16)sW[(long)k * V_ + v];  // soft_W is [E][V]
}

// Column sums of Wk/Wv (cross-attn kv is rank-1 because w is broadcast along E)
__global__ void k_csum(const float* __restrict__ Wk, const float* __restrict__ Wv,
                       float* __restrict__ csk, float* __restrict__ csv) {
  int tid = blockIdx.x * 256 + threadIdx.x;  // 2*O_*E_
  if (tid >= 2 * O_ * E_) return;
  int half = tid >> 9, i = (tid >> 8) & 1, e = tid & 255;
  const float* src = (half ? Wv : Wk) + (long)i * 65536;
  float s = 0.f;
  for (int in = 0; in < 256; ++in) s += src[in * 256 + e];
  (half ? csv : csk)[i * 256 + e] = s;
}

// w = noise @ W0 @ W1 @ W2 @ W3 (+biases); single block, LDS-resident
__global__ void k_wchain(const float* __restrict__ noise, const float* __restrict__ W,
                         const float* __restrict__ bia, float* __restrict__ wout) {
  __shared__ float w[B_][L_];
  int t = threadIdx.x;  // 1024
  int bb = t >> 6, j = t & 63;
  w[bb][j] = noise[bb * L_ + j];
  __syncthreads();
  for (int i = 0; i < 4; ++i) {
    float acc = bia[i * L_ + j];
    const float* Wi = W + i * L_ * L_;
    for (int k = 0; k < L_; ++k) acc += w[bb][k] * Wi[k * L_ + j];
    __syncthreads();
    w[bb][j] = acc;
    __syncthreads();
  }
  wout[bb * L_ + j] = w[bb][j];
}

__global__ void k_crosskv(const float* __restrict__ wc, const float* __restrict__ csk,
                          const float* __restrict__ csv, const float* __restrict__ bk,
                          const float* __restrict__ bv, float* __restrict__ Kc,
                          float* __restrict__ Vc) {
  int tid = blockIdx.x * 256 + threadIdx.x;  // O_*B_*L_*E_ = 524288
  if (tid >= O_ * B_ * L_ * E_) return;
  int i = tid >> 18, rem = tid & 262143;
  int bl = rem >> 8, e = rem & 255;
  float w = wc[bl];
  Kc[tid] = w * csk[i * 256 + e] + bk[i * 256 + e];
  Vc[tid] = w * csv[i * 256 + e] + bv[i * 256 + e];
}

// ---------------------------------------------------------------------------
// Per-step kernels
// ---------------------------------------------------------------------------
// Q (and for self blocks also K,V) projections. All 8 waves of a block share
// row0, so wave 0 TDM-stages the 16x256 bf16 A-slab into LDS once.
__global__ void k_qkv(const __bf16* __restrict__ Abf, const __bf16* __restrict__ WTq,
                      const __bf16* __restrict__ WTk, const __bf16* __restrict__ WTv,
                      const float* __restrict__ bq, const float* __restrict__ bk,
                      const float* __restrict__ bv, float* __restrict__ Qo,
                      float* __restrict__ Ko, float* __restrict__ Vo) {
  __shared__ __bf16 Alds[16 * ALDS_STRIDE];
  int wIn = threadIdx.x >> 5, lane = threadIdx.x & 31;
  int wid = blockIdx.x * 8 + wIn;
  int mat = wid >> 10;           // 1024 tiles per matrix
  int t = wid & 1023;
  int row0 = (t >> 4) << 4;      // uniform across the block
  int col0 = (t & 15) << 4;
  if (wIn == 0) {
    tdm_load_2d_bf16(lds_offset(Alds), Abf + (long)row0 * 256, 256, 16, 256);
    __builtin_amdgcn_s_wait_tensorcnt(0);
  }
  __syncthreads();
  const __bf16* WT = (mat == 0) ? WTq : ((mat == 1) ? WTk : WTv);
  const float* bias = (mat == 0) ? bq : ((mat == 1) ? bk : bv);
  float* Out = (mat == 0) ? Qo : ((mat == 1) ? Ko : Vo);
  f32x8 acc = wmma_gemm_lds(Alds, WT + (long)col0 * 256, lane);
  int m = lane & 15, hi = lane >> 4;
  float bb = bias[col0 + m];
#pragma unroll
  for (int r = 0; r < 8; ++r)
    Out[(long)(row0 + r + hi * 8) * 256 + col0 + m] = acc[r] + bb;
}

// One wave per (b,h,qrow); lane = head dim (DH==32==wave32). Mask matches ref (-1e9).
__global__ void k_attn(const float* __restrict__ Qb, const float* __restrict__ Kb,
                       const float* __restrict__ Vb, float* __restrict__ Ob,
                       __bf16* __restrict__ Obf, int tok) {
  __shared__ float sc[8][L_];
  int wIn = threadIdx.x >> 5, lane = threadIdx.x & 31;
  int wid = blockIdx.x * 8 + wIn;  // B_*H_*L_ = 8192 waves exactly
  int b = wid >> 9;
  int h = (wid >> 6) & 7;
  int qr = wid & 63;
  long baseQ = ((long)(b * L_ + qr)) * E_ + h * DH_;
  long kvRow = ((long)b * L_) * E_ + h * DH_;
  float qd = Qb[baseQ + lane] * 0.17677669529663687f;  // 1/sqrt(DH)
  for (int k = 0; k < L_; ++k) {
    float p = qd * Kb[kvRow + (long)k * E_ + lane];
#pragma unroll
    for (int mm = 16; mm; mm >>= 1) p += __shfl_xor(p, mm, 32);
    if (lane == 0) sc[wIn][k] = p;
  }
  float s0 = sc[wIn][lane];
  float s1 = sc[wIn][lane + 32];
  if (lane >= tok) s0 = -1e9f;
  if (lane + 32 >= tok) s1 = -1e9f;
  float mx = fmaxf(s0, s1);
#pragma unroll
  for (int mm = 16; mm; mm >>= 1) mx = fmaxf(mx, __shfl_xor(mx, mm, 32));
  float e0 = expf(s0 - mx), e1 = expf(s1 - mx);
  float sum = e0 + e1;
#pragma unroll
  for (int mm = 16; mm; mm >>= 1) sum += __shfl_xor(sum, mm, 32);
  float inv = 1.f / sum;
  sc[wIn][lane] = e0 * inv;
  sc[wIn][lane + 32] = e1 * inv;
  float acc = 0.f;
  for (int k = 0; k < L_; ++k)
    acc += sc[wIn][k] * Vb[kvRow + (long)k * E_ + lane];
  Ob[baseQ + lane] = acc;
  Obf[baseQ + lane] = (__bf16)acc;
}

// out-proj WMMA (A from TDM-staged LDS) + bias + residual (pre-LN)
__global__ void k_proj(const __bf16* __restrict__ Obf, const __bf16* __restrict__ WTo,
                       const float* __restrict__ bo, const float* __restrict__ Resid,
                       float* __restrict__ Out) {
  __shared__ __bf16 Alds[16 * ALDS_STRIDE];
  int wIn = threadIdx.x >> 5, lane = threadIdx.x & 31;
  int wid = blockIdx.x * 8 + wIn;  // 1024 waves exactly
  int row0 = (wid >> 4) << 4, col0 = (wid & 15) << 4;
  if (wIn == 0) {
    tdm_load_2d_bf16(lds_offset(Alds), Obf + (long)row0 * 256, 256, 16, 256);
    __builtin_amdgcn_s_wait_tensorcnt(0);
  }
  __syncthreads();
  f32x8 acc = wmma_gemm_lds(Alds, WTo + (long)col0 * 256, lane);
  int m = lane & 15, hi = lane >> 4;
  float bb = bo[col0 + m];
#pragma unroll
  for (int r = 0; r < 8; ++r) {
    long o = (long)(row0 + r + hi * 8) * 256 + col0 + m;
    Out[o] = acc[r] + bb + Resid[o];
  }
}

// LayerNorm per row; one wave per row. Writes f32 result + bf16 mirror.
__global__ void k_ln(const float* __restrict__ X, const float* __restrict__ g,
                     const float* __restrict__ bta, float* __restrict__ Out,
                     __bf16* __restrict__ Outbf) {
  int wIn = threadIdx.x >> 5, lane = threadIdx.x & 31;
  int row = blockIdx.x * 8 + wIn;  // 1024 rows exactly
  const float* x = X + (long)row * 256;
  float v[8];
  float s = 0.f;
#pragma unroll
  for (int j = 0; j < 8; ++j) { v[j] = x[j * 32 + lane]; s += v[j]; }
#pragma unroll
  for (int mm = 16; mm; mm >>= 1) s += __shfl_xor(s, mm, 32);
  float mu = s * (1.f / 256.f);
  float q = 0.f;
#pragma unroll
  for (int j = 0; j < 8; ++j) { float d = v[j] - mu; q += d * d; }
#pragma unroll
  for (int mm = 16; mm; mm >>= 1) q += __shfl_xor(q, mm, 32);
  float rstd = rsqrtf(q * (1.f / 256.f) + 1e-5f);
  float* o = Out + (long)row * 256;
  __bf16* ob = Outbf + (long)row * 256;
#pragma unroll
  for (int j = 0; j < 8; ++j) {
    int c = j * 32 + lane;
    float r = (v[j] - mu) * rstd * g[c] + bta[c];
    o[c] = r;
    ob[c] = (__bf16)r;
  }
}

__global__ void k_reset(unsigned long long* __restrict__ slots) {
  if (threadIdx.x < B_) slots[threadIdx.x] = 0ULL;
}

// monotonic pack: larger logit wins; tie -> smaller vocab index (like jnp.argmax)
__device__ __forceinline__ unsigned long long packmax(float f, int v) {
  unsigned u = __float_as_uint(f);
  u = (u & 0x80000000u) ? ~u : (u | 0x80000000u);
  return ((unsigned long long)u << 32) | (unsigned)(~v);
}

// logits: [B=16, E=256] x [E, V]. Batch rows are strided L_*E_ apart; TDM's 2-D
// descriptor gathers them into one LDS A-slab shared by all 8 waves.
__global__ void k_logits(const __bf16* __restrict__ Xbf, const __bf16* __restrict__ SWT,
                         const float* __restrict__ sb,
                         unsigned long long* __restrict__ slots, int tok) {
  __shared__ __bf16 Alds[16 * ALDS_STRIDE];
  int wIn = threadIdx.x >> 5, lane = threadIdx.x & 31;
  int wid = blockIdx.x * 8 + wIn;  // V_/16 = 1000 waves, 125 blocks exactly
  int col0 = wid << 4;
  if (wIn == 0) {
    tdm_load_2d_bf16(lds_offset(Alds), Xbf + (long)(tok - 1) * E_, 256, 16,
                     (unsigned long long)L_ * E_);
    __builtin_amdgcn_s_wait_tensorcnt(0);
  }
  __syncthreads();
  f32x8 acc = wmma_gemm_lds(Alds, SWT + (long)col0 * 256, lane);
  int m = lane & 15, hi = lane >> 4;
  int vcol = col0 + m;
  float sbv = sb[vcol];
#pragma unroll
  for (int r = 0; r < 8; ++r) {
    int b = r + hi * 8;
    unsigned long long p = packmax(acc[r] + sbv, vcol);
#pragma unroll
    for (int mm = 8; mm; mm >>= 1) {  // reduce within each 16-lane half
      unsigned long long o = __shfl_xor(p, mm, 32);
      if (o > p) p = o;
    }
    if (m == 0) atomicMax(&slots[b], p);
  }
}

__global__ void k_next(const unsigned long long* __restrict__ slots,
                       const float* __restrict__ emb, const float* __restrict__ PE,
                       float* __restrict__ Y, __bf16* __restrict__ Ybf,
                       int* __restrict__ out_tok, int tok) {
  __shared__ int sidx;
  int b = blockIdx.x;
  if (threadIdx.x == 0) {
    unsigned long long s = slots[b];
    int idx = (int)(~(unsigned)(s & 0xffffffffULL));
    sidx = idx;
    if (b == 0) out_tok[tok] = idx;
  }
  __syncthreads();
  int idx = sidx;
  int e = threadIdx.x;  // 256 threads
  float v = emb[(long)idx * E_ + e] + PE[(long)tok * E_ + e];
  long o = ((long)b * L_ + tok) * E_ + e;
  Y[o] = v;
  Ybf[o] = (__bf16)v;
}

// ---------------------------------------------------------------------------
extern "C" void kernel_launch(void* const* d_in, const int* in_sizes, int n_in,
                              void* d_out, int out_size, void* d_ws, size_t ws_size,
                              hipStream_t stream) {
  const float* noise = (const float*)d_in[0];
  const float* inW   = (const float*)d_in[1];
  const float* inB   = (const float*)d_in[2];
  const float* Wq    = (const float*)d_in[3];
  const float* bq    = (const float*)d_in[4];
  const float* Wk    = (const float*)d_in[5];
  const float* bk    = (const float*)d_in[6];
  const float* Wv    = (const float*)d_in[7];
  const float* bv    = (const float*)d_in[8];
  const float* Wo    = (const float*)d_in[9];
  const float* bo    = (const float*)d_in[10];
  const float* lng   = (const float*)d_in[11];
  const float* lnb   = (const float*)d_in[12];
  const float* emb   = (const float*)d_in[13];
  const float* sW    = (const float*)d_in[14];
  const float* sb    = (const float*)d_in[15];
  int* out_tok = (int*)d_out;

  char* ws = (char*)d_ws;
  size_t off = 0;
  auto carve = [&](size_t bytes) -> char* {
    char* p = ws + off;
    off = (off + bytes + 255) & ~(size_t)255;
    return p;
  };
  const size_t ACT  = (size_t)B_ * L_ * E_ * 4;  // 1 MB f32 activation
  const size_t ACTB = (size_t)B_ * L_ * E_ * 2;  // bf16 mirror
  float*  PEd    = (float*)carve((size_t)L_ * E_ * 4);
  float*  wch    = (float*)carve((size_t)B_ * L_ * 4);
  float*  Y      = (float*)carve(ACT);
  float*  bufA   = (float*)carve(ACT);
  float*  bufB   = (float*)carve(ACT);
  float*  Qb     = (float*)carve(ACT);
  float*  Obuf   = (float*)carve(ACT);
  float*  Tmp    = (float*)carve(ACT);
  float*  Ks     = (float*)carve(ACT);
  float*  Vs     = (float*)carve(ACT);
  float*  Kc     = (float*)carve((size_t)O_ * ACT);
  float*  Vc     = (float*)carve((size_t)O_ * ACT);
  float*  csk    = (float*)carve((size_t)O_ * E_ * 4);
  float*  csv    = (float*)carve((size_t)O_ * E_ * 4);
  __bf16* WT     = (__bf16*)carve((size_t)8 * 65536 * 2);
  __bf16* SWT    = (__bf16*)carve((size_t)V_ * E_ * 2);
  __bf16* Ybf    = (__bf16*)carve(ACTB);
  __bf16* bufAbf = (__bf16*)carve(ACTB);
  __bf16* bufBbf = (__bf16*)carve(ACTB);
  __bf16* Obf    = (__bf16*)carve(ACTB);
  unsigned long long* slots = (unsigned long long*)carve(B_ * 8);

  dim3 blk(256);
  // ---- one-time prep (re-run every call: deterministic, ws may be poisoned)
  k_prep_pe_y<<<(B_ * L_ * E_ + 255) / 256, blk, 0, stream>>>(PEd, Y, Ybf, out_tok);
  k_y0<<<(B_ * E_ + 255) / 256, blk, 0, stream>>>(Y, Ybf, PEd, emb);
  k_trsq<<<(8 * 65536) / 256, blk, 0, stream>>>(Wq, Wk, Wv, Wo, WT);
  k_trsoft<<<(int)(((long)V_ * E_) / 256), blk, 0, stream>>>(sW, SWT);
  k_csum<<<(2 * O_ * E_ + 255) / 256, blk, 0, stream>>>(Wk, Wv, csk, csv);
  k_wchain<<<1, 1024, 0, stream>>>(noise, inW, inB, wch);
  k_crosskv<<<(O_ * B_ * L_ * E_) / 256, blk, 0, stream>>>(wch, csk, csv, bk, bv, Kc, Vc);

  // ---- 63 sequential decode steps
  for (int tok = 1; tok < L_; ++tok) {
    const float*  cur   = Y;
    const __bf16* curbf = Ybf;
    for (int j = 0; j < 4; ++j) {  // blocks 0,1 cross (kv=w rank-1); 2,3 self
      int pi = j & 1;
      float*  outb   = (j == 0 || j == 2) ? bufA : bufB;
      __bf16* outbbf = (j == 0 || j == 2) ? bufAbf : bufBbf;
      const __bf16* WTq = WT + (long)(0 * 2 + pi) * 65536;
      const __bf16* WTk = WT + (long)(1 * 2 + pi) * 65536;
      const __bf16* WTv = WT + (long)(2 * 2 + pi) * 65536;
      const __bf16* WTo = WT + (long)(3 * 2 + pi) * 65536;
      int nm = (j < 2) ? 1 : 3;  // cross blocks need only Q
      k_qkv<<<nm * 128, blk, 0, stream>>>(curbf, WTq, WTk, WTv, bq + pi * E_,
                                          bk + pi * E_, bv + pi * E_, Qb, Ks, Vs);
      const float* Kp = (j < 2) ? (Kc + (long)pi * B_ * L_ * E_) : Ks;
      const float* Vp = (j < 2) ? (Vc + (long)pi * B_ * L_ * E_) : Vs;
      k_attn<<<1024, blk, 0, stream>>>(Qb, Kp, Vp, Obuf, Obf, tok);
      k_proj<<<128, blk, 0, stream>>>(Obf, WTo, bo + pi * E_, cur, Tmp);
      k_ln<<<128, blk, 0, stream>>>(Tmp, lng + pi * E_, lnb + pi * E_, outb, outbbf);
      cur = outb;
      curbf = outbbf;
    }
    k_reset<<<1, 32, 0, stream>>>(slots);
    k_logits<<<V_ / 16 / 8, blk, 0, stream>>>(curbf, SWT, sb, slots, tok);
    k_next<<<B_, 256, 0, stream>>>(slots, emb, PEd, Y, Ybf, out_tok, tok);
  }
}
